// MCL_22625887715478
// MI455X (gfx1250) — compile-verified, matrix operand order checked by reading, not run on hardware
//
#include <hip/hip_runtime.h>
#include <hip/hip_bf16.h>

// Problem constants (match reference)
#define BB 128
#define CC 1024
#define NWORDS 32          // CC / 32 bits per word
#define SLICES 4           // i-dimension split per batch row
#define TPB 256            // threads per main block (8 wave32s)

typedef __attribute__((ext_vector_type(16))) _Float16 v16h;
typedef __attribute__((ext_vector_type(8)))  float    v8f;

// ---------------------------------------------------------------------------
// Wave-level sum of n floats using WMMA "ones-matmul" reduction:
//   D = ones(16x32,f16) x B(32x16,f16) + C(f32)
// Every one of the 512 B-elements per call lands in exactly one (k,n) slot,
// and with all-ones A each D row equals the per-column sum, so
// sum_n D[0,n] == sum of all packed values (layout-independent).
// Must be called by a fully-active wave32 (EXEC all ones).
// ---------------------------------------------------------------------------
__device__ __forceinline__ float wave_sum(const float* __restrict__ buf, int n, int lane) {
#if defined(__gfx1250__) && __has_builtin(__builtin_amdgcn_wmma_f32_16x16x32_f16)
    v16h ones;
#pragma unroll
    for (int r = 0; r < 16; ++r) ones[r] = (_Float16)1.0f;
    v8f acc = {0.f, 0.f, 0.f, 0.f, 0.f, 0.f, 0.f, 0.f};
    const int per = 512;                       // values consumed per WMMA
    const int calls = (n + per - 1) / per;
    for (int t = 0; t < calls; ++t) {
        v16h bh;
#pragma unroll
        for (int r = 0; r < 16; ++r) {
            int idx = t * per + r * 32 + lane; // stride-32: conflict-free LDS / coalesced global
            bh[r] = (idx < n) ? (_Float16)buf[idx] : (_Float16)0.0f;
        }
        // 8 args: (neg_a, A, neg_b, B, c_mod, C, reuse_a, reuse_b)
        acc = __builtin_amdgcn_wmma_f32_16x16x32_f16(
            false, ones, false, bh, (short)0, acc, false, false);
    }
    // acc[0]: lanes 0-15 hold D[0,N]; lanes 16-31 hold D[8,N] (identical rows).
    // XOR-reduce within groups of 16 -> lane 0 holds sum_n D[0,n] == total.
    float v = acc[0];
#pragma unroll
    for (int m = 1; m < 16; m <<= 1) v += __shfl_xor(v, m, 32);
    return v;
#else
    // Scalar fallback (also what the host-side parse sees).
    float s = 0.0f;
    for (int k = lane; k < n; k += 32) s += buf[k];
    for (int m = 1; m < 32; m <<= 1) s += __shfl_xor(s, m, 32);
    return s;
#endif
}

// ---------------------------------------------------------------------------
// Kernel 1: bit-pack the binary float mask (4 MB) into 128 KB of bits.
// Layout: packed[word*CC + i], bit jj = mask[i][word*32 + jj].
// Wave-aligned lanes cover 32 consecutive j -> one __ballot == one word.
// ---------------------------------------------------------------------------
__global__ __launch_bounds__(256) void MCL_asl_bitpack(const float* __restrict__ mask,
                                                       unsigned* __restrict__ packed) {
    unsigned g = blockIdx.x * 256u + threadIdx.x;   // g in [0, CC*CC)
    unsigned i = g >> 10;                           // mask row
    unsigned j = g & 1023u;                         // mask col
    float m = mask[g];
    unsigned long long bal = __ballot(m > 0.5f);    // wave32: low 32 bits valid
    if ((j & 31u) == 0u) {
        packed[(j >> 5) * CC + i] = (unsigned)bal;
    }
}

// ---------------------------------------------------------------------------
// Kernel 2: per (b, i-slice) block — masked max over j, ASL loss terms,
// WMMA block reduction -> partial sums (unscaled, so f16 stays O(1..1e3)).
// ---------------------------------------------------------------------------
__global__ __launch_bounds__(TPB) void MCL_asl_main(const float* __restrict__ x,
                                                    const float* __restrict__ y,
                                                    const unsigned* __restrict__ packed,
                                                    float* __restrict__ partials) {
    __shared__ __align__(16) float w[CC];     // w[j] = x[b,j]^2 * y[b,j]
    __shared__ __align__(16) float red[TPB];

    const int b     = blockIdx.x;
    const int slice = blockIdx.y;
    const int tid   = threadIdx.x;
    const int i     = slice * TPB + tid;

    // Cooperative, vectorized build of w[b, :] in LDS (each thread: 4 j's).
    {
        const float4* x4 = reinterpret_cast<const float4*>(x + b * CC);
        const float4* y4 = reinterpret_cast<const float4*>(y + b * CC);
        float4 xv = x4[tid];
        float4 yv = y4[tid];
        float4 wv;
        wv.x = xv.x * xv.x * yv.x;
        wv.y = xv.y * xv.y * yv.y;
        wv.z = xv.z * xv.z * yv.z;
        wv.w = xv.w * xv.w * yv.w;
        reinterpret_cast<float4*>(w)[tid] = wv;
    }
    __syncthreads();

    // Masked max over all j for this thread's row i.
    // packed[word*CC + i]: lanes have consecutive i -> perfectly coalesced.
    // w reads are same-address broadcasts (no bank conflicts), b128 loads.
    float mmax = 0.0f;
#pragma unroll 2
    for (int word = 0; word < NWORDS; ++word) {
        unsigned bits = packed[word * CC + i];
        const float4* wp4 = reinterpret_cast<const float4*>(w + (word << 5));
#pragma unroll
        for (int q = 0; q < 8; ++q) {
            float4 wv = wp4[q];
            unsigned bq = bits >> (q * 4);
            mmax = fmaxf(mmax, (bq & 1u) ? wv.x : 0.0f);
            mmax = fmaxf(mmax, (bq & 2u) ? wv.y : 0.0f);
            mmax = fmaxf(mmax, (bq & 4u) ? wv.z : 0.0f);
            mmax = fmaxf(mmax, (bq & 8u) ? wv.w : 0.0f);
        }
    }

    // ASL focal loss (GAMMA_POS = 0 -> pow(1-xp,0) == 1; GAMMA_NEG = 4).
    const float xi = x[b * CC + i];
    const float yi = y[b * CC + i];
    const float xp  = (1.0f - yi) * xi + yi * mmax;
    const float lp  = yi * logf(xp);
    const float xp2 = xp * xp;
    const float ln  = (1.0f - yi) * (xp2 * xp2) * logf(1.0f - xp);
    red[tid] = -(lp + ln);                // scaling by 1/(B*C) deferred to the end
    __syncthreads();

    if (tid < 32) {                       // wave 0, fully active -> EXEC all ones
        float s = wave_sum(red, TPB, tid);
        if (tid == 0) partials[b * SLICES + slice] = s;
    }
}

// ---------------------------------------------------------------------------
// Kernel 3: single wave reduces the 512 block partials via WMMA, applies the
// mean scale, overwrites d_out[0] (deterministic across graph replays).
// ---------------------------------------------------------------------------
__global__ __launch_bounds__(32) void MCL_asl_final(const float* __restrict__ partials,
                                                    float* __restrict__ out) {
    const int lane = threadIdx.x;
    float total = wave_sum(partials, BB * SLICES, lane);
    if (lane == 0) {
        out[0] = total * (1.0f / (float)(BB * CC));
    }
}

// ---------------------------------------------------------------------------
extern "C" void kernel_launch(void* const* d_in, const int* in_sizes, int n_in,
                              void* d_out, int out_size, void* d_ws, size_t ws_size,
                              hipStream_t stream) {
    (void)in_sizes; (void)n_in; (void)out_size; (void)ws_size;
    const float* x    = (const float*)d_in[0];   // [B, C] f32
    const float* y    = (const float*)d_in[1];   // [B, C] f32
    const float* mask = (const float*)d_in[2];   // [C, C] f32

    // Workspace layout: [0, 128KB) packed mask bits; then 512 partial sums.
    unsigned* packed  = (unsigned*)d_ws;
    float* partials   = (float*)((char*)d_ws + (size_t)NWORDS * CC * sizeof(unsigned));
    float* out        = (float*)d_out;

    MCL_asl_bitpack<<<(CC * CC) / 256, 256, 0, stream>>>(mask, packed);

    dim3 grid(BB, SLICES);
    MCL_asl_main<<<grid, TPB, 0, stream>>>(x, y, packed, partials);

    MCL_asl_final<<<1, 32, 0, stream>>>(partials, out);
}
